// _ModConv_87368224735489
// MI455X (gfx1250) — matrix operand order repaired and checked by reading, not compiled, not run on hardware
//
#include <hip/hip_runtime.h>

// ---------------------------------------------------------------------------
// Modulated conv1d (StyleGAN-style), B=16, I=128, O=256, K=3, T=256, L=8192.
// Strategy: fold the 3-tap conv into a per-batch GEMM over K-dim = I*K = 384
// and run it on v_wmma_f32_16x16x32_bf16 (wave32, CDNA5).
// ---------------------------------------------------------------------------

typedef __attribute__((ext_vector_type(16))) __bf16 v16bf;
typedef __attribute__((ext_vector_type(8)))  __bf16 v8bf;
typedef __attribute__((ext_vector_type(8)))  float  v8f;

#define B_ 16
#define I_ 128
#define O_ 256
#define K_ 3
#define T_ 256
#define L_ 8192

__device__ __forceinline__ unsigned short f2bf(float f) {
    // round-to-nearest-even fp32 -> bf16
    unsigned u = __float_as_uint(f);
    u += 0x7FFFu + ((u >> 16) & 1u);
    return (unsigned short)(u >> 16);
}

// ---------------------------------------------------------------------------
// Kernel 0: style vector  s[b][i] = t[b] . mod_w[i] + mod_b[i] + 1
// ---------------------------------------------------------------------------
__global__ __launch_bounds__(128) void style_kernel(
    const float* __restrict__ t, const float* __restrict__ mod_w,
    const float* __restrict__ mod_b, float* __restrict__ s)
{
    const int b = blockIdx.x;
    const int i = threadIdx.x;
    float acc = mod_b[i] + 1.0f;
    for (int tt = 0; tt < T_; ++tt)
        acc += t[b * T_ + tt] * mod_w[i * T_ + tt];
    s[b * I_ + i] = acc;
}

// ---------------------------------------------------------------------------
// Kernel 1: modulate + demodulate, pack bf16 weights as wb[b][k][o][i]
// (row o contiguous in i -> 16B-aligned A-fragment loads)
// ---------------------------------------------------------------------------
__global__ __launch_bounds__(128) void modulate_kernel(
    const float* __restrict__ conv_w, const float* __restrict__ s,
    unsigned short* __restrict__ wb)
{
    __shared__ float red[128];
    const int bo = blockIdx.x;
    const int b  = bo >> 8;        // / O_
    const int o  = bo & (O_ - 1);  // % O_
    const int i  = threadIdx.x;

    const float sv = s[b * I_ + i];
    float wv[3];
    float p = 0.0f;
#pragma unroll
    for (int k = 0; k < K_; ++k) {
        wv[k] = conv_w[((size_t)o * I_ + i) * K_ + k] * sv;
        p += wv[k] * wv[k];
    }
    red[i] = p;
    __syncthreads();
    for (int st = 64; st > 0; st >>= 1) {
        if (i < st) red[i] += red[i + st];
        __syncthreads();
    }
    const float d = __frsqrt_rn(red[0] + 1e-8f);
#pragma unroll
    for (int k = 0; k < K_; ++k)
        wb[((size_t)(b * K_ + k) * O_ + o) * I_ + i] = f2bf(wv[k] * d);
}

// ---------------------------------------------------------------------------
// Kernel 2: main GEMM-conv.
// grid (L/64, 1, B), 256 thr = 8 waves. Block output tile: 256(O) x 64(L).
// Each wave: 2 M-tiles x 4 N-tiles of 16x16 f32 accum; K loop = 12 x 32.
// x tile staged once to LDS as bf16, transposed [col][i], col = l - l0 + 1,
// padded stride 136 shorts to stagger banks. Tap k selects LDS column c+k.
// ---------------------------------------------------------------------------
__global__ __launch_bounds__(256) void modconv_wmma_kernel(
    const float* __restrict__ x, const unsigned short* __restrict__ wb,
    float* __restrict__ out)
{
    constexpr int NT  = 64;       // L columns per block
    constexpr int NC  = NT + 2;   // staged columns (halo for 3-tap, pad=1)
    constexpr int STR = 136;      // LDS column stride in shorts (bank stagger)
    __shared__ __align__(16) unsigned short xT[NC * STR];

    const int b    = blockIdx.z;
    const int l0   = blockIdx.x * NT;
    const int tid  = threadIdx.x;
    const int lane = tid & 31;
    const int w    = tid >> 5;

    // ---- stage x tile: fp32 -> bf16, transposed into LDS -----------------
    const float* xb = x + (size_t)b * I_ * L_;
    for (int idx = tid; idx < I_ * NC; idx += 256) {
        const int i = idx / NC;
        const int c = idx % NC;
        const int l = l0 - 1 + c;
        const float v = (l >= 0 && l < L_) ? xb[(size_t)i * L_ + l] : 0.0f;
        xT[c * STR + i] = f2bf(v);
    }
    __syncthreads();

    // ---- accumulators -----------------------------------------------------
    v8f acc[2][4];
#pragma unroll
    for (int m = 0; m < 2; ++m)
#pragma unroll
        for (int n = 0; n < 4; ++n) acc[m][n] = v8f{0.f,0.f,0.f,0.f,0.f,0.f,0.f,0.f};

    const int mrow = lane & 15;
    const int hiA  = (lane >> 4) ? 16 : 0;   // A: K cols 0-15 / 16-31
    const int hiB  = (lane >> 4) ? 16 : 0;   // B: K rows 0-15 / 16-31

    // ---- K loop: 12 steps of 32 over (tap k, input channel i) -------------
#pragma unroll
    for (int kk = 0; kk < 12; ++kk) {
        const int ks = kk >> 2;         // tap 0..2
        const int i0 = (kk & 3) * 32;   // channel block

        // A fragments: modulated weights (bf16, row-major O x I per tap)
        v16bf afrag[2];
#pragma unroll
        for (int m = 0; m < 2; ++m) {
            const int o = (w + m * 8) * 16 + mrow;
            const unsigned short* ap =
                wb + ((size_t)(b * K_ + ks) * O_ + o) * I_ + i0 + hiA;
            const v8bf lo = *reinterpret_cast<const v8bf*>(ap);
            const v8bf hi = *reinterpret_cast<const v8bf*>(ap + 8);
            afrag[m] = __builtin_shufflevector(lo, hi,
                0,1,2,3,4,5,6,7,8,9,10,11,12,13,14,15);
        }

        // B fragments from LDS (column = output l + tap shift)
#pragma unroll
        for (int n = 0; n < 4; ++n) {
            const int c = n * 16 + mrow + ks;
            const unsigned short* bp = &xT[c * STR + i0 + hiB];
            const v8bf lo = *reinterpret_cast<const v8bf*>(bp);
            const v8bf hi = *reinterpret_cast<const v8bf*>(bp + 8);
            const v16bf bfrag = __builtin_shufflevector(lo, hi,
                0,1,2,3,4,5,6,7,8,9,10,11,12,13,14,15);
#pragma unroll
            for (int m = 0; m < 2; ++m)
                acc[m][n] = __builtin_amdgcn_wmma_f32_16x16x32_bf16(
                    false, afrag[m], false, bfrag,
                    (short)0, acc[m][n], false, false);
        }
    }

    // ---- store D (16x16 f32 layout: VGPR j -> M=j(+8 hi half), N=lane&15) -
    float* ob = out + (size_t)b * O_ * L_;
#pragma unroll
    for (int m = 0; m < 2; ++m) {
        const int obase = (w + m * 8) * 16 + ((lane >> 4) << 3);
#pragma unroll
        for (int n = 0; n < 4; ++n) {
            const int l = l0 + n * 16 + (lane & 15);
#pragma unroll
            for (int j = 0; j < 8; ++j)
                ob[(size_t)(obase + j) * L_ + l] = acc[m][n][j];
        }
    }
}

// ---------------------------------------------------------------------------
// Launch
// ---------------------------------------------------------------------------
extern "C" void kernel_launch(void* const* d_in, const int* in_sizes, int n_in,
                              void* d_out, int out_size, void* d_ws, size_t ws_size,
                              hipStream_t stream) {
    const float* x      = (const float*)d_in[0];  // [B, I, L]
    const float* t      = (const float*)d_in[1];  // [B, T]
    const float* conv_w = (const float*)d_in[2];  // [O, I, K]
    const float* mod_w  = (const float*)d_in[3];  // [I, T]
    const float* mod_b  = (const float*)d_in[4];  // [I]
    float* out = (float*)d_out;                   // [B, O, L]

    // workspace: s [B*I] f32 at offset 0; packed bf16 weights after 8KB
    float* s_ws = (float*)d_ws;
    unsigned short* wb = (unsigned short*)((char*)d_ws + 8192);

    style_kernel<<<dim3(B_), 128, 0, stream>>>(t, mod_w, mod_b, s_ws);
    modulate_kernel<<<dim3(B_ * O_), 128, 0, stream>>>(conv_w, s_ws, wb);
    modconv_wmma_kernel<<<dim3(L_ / 64, 1, B_), 256, 0, stream>>>(x, wb, out);
}
